// GPSLayer_77936476553894
// MI455X (gfx1250) — compile-verified, matrix operand order checked by reading, not run on hardware
//
#include <hip/hip_runtime.h>
#include <hip/hip_bf16.h>
#include <stdint.h>

// ---------------------------------------------------------------------------
// CDNA5 (gfx1250) GPS layer: bf16 WMMA GEMMs + async-LDS copies + flash attn.
// ---------------------------------------------------------------------------

#define USE_ASYNC_COPY 1

typedef __bf16  v16bf __attribute__((ext_vector_type(16)));
typedef float   v8f   __attribute__((ext_vector_type(8)));

static __device__ __forceinline__ v8f wmma_bf16(v16bf a, v16bf b, v8f c) {
    return __builtin_amdgcn_wmma_f32_16x16x32_bf16(
        false, a, false, b, (short)0, c, false, false);
}

// Fragment layout (ISA 16-bit A/B 16x32): lanes 0-15: row=lane, K in
// {0..7,16..23}; lanes 16-31: row=lane-16, K in {8..15,24..31}.
// Rows are 32 halfs = four 16B chunks; a lane needs chunks {lh, 2+lh}.
static __device__ __forceinline__ v16bf load_frag16(const __bf16* base, int lane) {
    const int r  = lane & 15;
    const int lh = lane >> 4;
    const uint4* p = (const uint4*)(base + r * 32);
    union { uint4 q[2]; v16bf f; } u;
    u.q[0] = p[lh];
    u.q[1] = p[2 + lh];
    return u.f;
}

static __device__ __forceinline__ uint32_t lds_off(const void* p) {
    // gfx1250 flat->LDS mapping uses addr[31:0] as the LDS byte offset.
    return (uint32_t)(uintptr_t)p;
}

static __device__ __forceinline__ void async_copy16(uint32_t lds, const __bf16* g) {
#if USE_ASYNC_COPY
    asm volatile("global_load_async_to_lds_b128 %0, %1, off"
                 :: "v"(lds), "v"((uint64_t)(uintptr_t)g) : "memory");
#else
    (void)lds; (void)g;
#endif
}

static __device__ __forceinline__ void wait_async0() {
#if USE_ASYNC_COPY
    asm volatile("s_wait_asynccnt 0x0" ::: "memory");
#endif
}

// ---------------------------------------------------------------------------
// NT GEMM on bf16:  C[M,N] = epilogue(A[M,K] @ W[N,K]^T + bias)
// MODE 0: f32 out (+bias)         MODE 1: bf16 out, relu(+bias)
// MODE 2: bf16 out, (+bias)*scale MODE 3: f32 out, distance-bias epilogue
// N, K compile-time so all row offsets fold to immediates.
// Block 256 threads (8 waves). Tile BM=128, BN=64, BK=32; wave tile 32x32.
// Double-buffered LDS fed by GLOBAL_LOAD_ASYNC_TO_LDS_B128 (ASYNCcnt).
// ---------------------------------------------------------------------------
template <int MODE, int N, int K>
__global__ __launch_bounds__(256)
void gemm_nt(const __bf16* __restrict__ A, const __bf16* __restrict__ W,
             const float* __restrict__ bias, float* __restrict__ Cf,
             __bf16* __restrict__ Cb, const float* __restrict__ a2,
             float scale, long strideA, long strideW, long strideC) {
    constexpr int BM = 128, BN = 64, BK = 32;
    __shared__ __align__(16) __bf16 As[2][BM][BK];
    __shared__ __align__(16) __bf16 Ws[2][BN][BK];

    A += (size_t)blockIdx.z * strideA;
    W += (size_t)blockIdx.z * strideW;

    const int m0   = blockIdx.x * BM;
    const int n0   = blockIdx.y * BN;
    const int tid  = threadIdx.x;
    const int lane = tid & 31;
    const int w    = tid >> 5;
    const int wm   = (w >> 1) * 32;   // 4 waves along M
    const int wn   = (w & 1) * 32;    // 2 waves along N

    auto issue = [&](int buf, int k0) {
#if USE_ASYNC_COPY
#pragma unroll
        for (int t = 0; t < 2; ++t) {            // A: 512 chunks of 16B
            int ch = tid + t * 256;
            int r  = ch >> 2, c = (ch & 3) * 8;
            async_copy16(lds_off(&As[buf][r][c]),
                         A + (size_t)(m0 + r) * K + k0 + c);
        }
        {                                         // W: 256 chunks of 16B
            int r = tid >> 2, c = (tid & 3) * 8;
            async_copy16(lds_off(&Ws[buf][r][c]),
                         W + (size_t)(n0 + r) * K + k0 + c);
        }
#else
#pragma unroll
        for (int t = 0; t < 2; ++t) {
            int ch = tid + t * 256;
            int r  = ch >> 2, c = (ch & 3) * 8;
            *(uint4*)&As[buf][r][c] =
                *(const uint4*)(A + (size_t)(m0 + r) * K + k0 + c);
        }
        {
            int r = tid >> 2, c = (tid & 3) * 8;
            *(uint4*)&Ws[buf][r][c] =
                *(const uint4*)(W + (size_t)(n0 + r) * K + k0 + c);
        }
#endif
    };

    v8f acc[2][2] = {};
    int cur = 0;
    issue(0, 0);
#if !USE_ASYNC_COPY
    __syncthreads();
#endif

    for (int k0 = 0; k0 < K; k0 += BK) {
        wait_async0();
        __syncthreads();
        if (k0 + BK < K) issue(cur ^ 1, k0 + BK);

        v16bf a0 = load_frag16(&As[cur][wm][0],      lane);
        v16bf a1 = load_frag16(&As[cur][wm + 16][0], lane);
        v16bf b0 = load_frag16(&Ws[cur][wn][0],      lane);
        v16bf b1 = load_frag16(&Ws[cur][wn + 16][0], lane);

        acc[0][0] = wmma_bf16(a0, b0, acc[0][0]);
        acc[0][1] = wmma_bf16(a0, b1, acc[0][1]);
        acc[1][0] = wmma_bf16(a1, b0, acc[1][0]);
        acc[1][1] = wmma_bf16(a1, b1, acc[1][1]);

        cur ^= 1;
#if !USE_ASYNC_COPY
        __syncthreads();
#endif
    }

    // Epilogue. C layout: VGPR i, lanes0-15 -> M=i, N=lane; lanes16-31 -> M=8+i.
    const int ln = lane & 15;
    const int lm = (lane >> 4) * 8;
#pragma unroll
    for (int ai = 0; ai < 2; ++ai) {
#pragma unroll
        for (int bi = 0; bi < 2; ++bi) {
            const int row0 = m0 + wm + ai * 16 + lm;
            const int col  = n0 + wn + bi * 16 + ln;
            const size_t base =
                (size_t)blockIdx.z * strideC + (size_t)row0 * N + col;
#pragma unroll
            for (int i = 0; i < 8; ++i) {
                const size_t idx = base + (size_t)i * N;   // immediate offsets
                float v = acc[ai][bi][i];
                if (MODE == 0) {
                    Cf[idx] = v + bias[col];
                } else if (MODE == 1) {
                    float t = v + bias[col];
                    Cb[idx] = (__bf16)(t > 0.f ? t : 0.f);
                } else if (MODE == 2) {
                    Cb[idx] = (__bf16)((v + bias[col]) * scale);
                } else { // 2 + 1e-24 - sqrt(max(|a|^2+|b|^2-2ab, 0))
                    const float* a2b = a2 + (size_t)blockIdx.z * N;
                    float d2 = a2b[row0 + i] + a2b[col] - 2.0f * v;
                    Cf[idx]  = 2.0f + 1e-24f - sqrtf(fmaxf(d2, 0.0f));
                }
            }
        }
    }
}

// ---------------------------------------------------------------------------
// Flash attention: one wave per 16-row tile of one (batch, head). hd = 32.
// S = Q K^T + bias, online softmax, O += P V. Output bf16 (feeds W_O GEMM).
// All dims compile-time; streaming pointers strength-reduced.
// ---------------------------------------------------------------------------
template <int N, int Dm, int H, int hd>
__global__ __launch_bounds__(32)
void flash_attn(const __bf16* __restrict__ q, const __bf16* __restrict__ k,
                const __bf16* __restrict__ v, const float* __restrict__ bias,
                __bf16* __restrict__ out) {
    const int bh = blockIdx.y;
    const int b  = bh / H;
    const int h  = bh % H;
    const int r0 = blockIdx.x * 16;
    const int lane = threadIdx.x;
    const int lr = lane & 15;
    const int lh = lane >> 4;
    const int kb = lh * 8;

    __shared__ __align__(16) __bf16 Pt[16 * 32];
    __shared__ __align__(16) __bf16 Vs[32 * 32];

    // Q fragment: two 16B chunks of the (row, head) slice.
    v16bf qf;
    {
        const uint4* qp = (const uint4*)(q + ((size_t)(b * N + r0 + lr)) * Dm + h * hd);
        union { uint4 qq[2]; v16bf f; } u;
        u.qq[0] = qp[lh];
        u.qq[1] = qp[2 + lh];
        qf = u.f;
    }

    v8f acc0 = {}, acc1 = {};
    float m8[8], l8[8];
#pragma unroll
    for (int i = 0; i < 8; ++i) { m8[i] = -1e30f; l8[i] = 0.f; }

    // Streaming pointers (advance per 32-col tile).
    const __bf16* kp = k + ((size_t)(b * N + lr)) * Dm + h * hd;
    const __bf16* vp = v + ((size_t)(b * N + lane)) * Dm + h * hd;
    const float*  bp = bias + ((size_t)(b * N + r0 + lh * 8)) * N + lr;
    const v8f zero = {};

    for (int j0 = 0; j0 < N; j0 += 32) {
        __builtin_prefetch(kp + (size_t)32 * Dm, 0, 1);
        v16bf kf0, kf1;
        {
            const uint4* k0p = (const uint4*)kp;
            const uint4* k1p = (const uint4*)(kp + (size_t)16 * Dm);
            union { uint4 qq[2]; v16bf f; } u0, u1;
            u0.qq[0] = k0p[lh]; u0.qq[1] = k0p[2 + lh];
            u1.qq[0] = k1p[lh]; u1.qq[1] = k1p[2 + lh];
            kf0 = u0.f; kf1 = u1.f;
        }
        v8f s0 = wmma_bf16(qf, kf0, zero);
        v8f s1 = wmma_bf16(qf, kf1, zero);

        // bias + online softmax (row m = lh*8+i, shared by the 16 lanes with
        // the same lane>>4 bit; xor masks 1..8 never cross the half-wave).
#pragma unroll
        for (int i = 0; i < 8; ++i) {
            float x0 = s0[i] + bp[(size_t)i * N];        // immediate offsets
            float x1 = s1[i] + bp[(size_t)i * N + 16];
            float mx = fmaxf(x0, x1);
#pragma unroll
            for (int off = 1; off < 16; off <<= 1)
                mx = fmaxf(mx, __shfl_xor(mx, off, 32));
            float mn   = fmaxf(m8[i], mx);
            float corr = __expf(m8[i] - mn);
            float p0 = __expf(x0 - mn);
            float p1 = __expf(x1 - mn);
            float ps = p0 + p1;
#pragma unroll
            for (int off = 1; off < 16; off <<= 1)
                ps += __shfl_xor(ps, off, 32);
            l8[i] = l8[i] * corr + ps;
            m8[i] = mn;
            acc0[i] *= corr;
            acc1[i] *= corr;
            Pt[(lh * 8 + i) * 32 + lr]      = (__bf16)p0;
            Pt[(lh * 8 + i) * 32 + 16 + lr] = (__bf16)p1;
        }

        // Stage V tile: lane loads one 32-half row (64B).
        {
            uint4*       d = (uint4*)&Vs[lane * 32];
            const uint4* s = (const uint4*)vp;
            d[0] = s[0]; d[1] = s[1]; d[2] = s[2]; d[3] = s[3];
        }

        // O += P V  (per-wave DS ops are in-order, so no barrier needed)
        v16bf pf = load_frag16(Pt, lane);
        v16bf vf0, vf1;
#pragma unroll
        for (int i = 0; i < 8; ++i) {
            vf0[i]     = Vs[(kb + i) * 32 + lr];
            vf0[8 + i] = Vs[(16 + kb + i) * 32 + lr];
            vf1[i]     = Vs[(kb + i) * 32 + 16 + lr];
            vf1[8 + i] = Vs[(16 + kb + i) * 32 + 16 + lr];
        }
        acc0 = wmma_bf16(pf, vf0, acc0);
        acc1 = wmma_bf16(pf, vf1, acc1);

        kp += (size_t)32 * Dm;
        vp += (size_t)32 * Dm;
        bp += 32;
    }

    // Normalize and write O (bf16 -> feeds W_O GEMM).
#pragma unroll
    for (int i = 0; i < 8; ++i) {
        const int m = lh * 8 + i;
        const float inv = 1.0f / l8[i];
        const size_t o = ((size_t)(b * N + r0 + m)) * Dm + h * hd;
        out[o + lr]      = (__bf16)(acc0[i] * inv);
        out[o + 16 + lr] = (__bf16)(acc1[i] * inv);
    }
}

// ---------------------------------------------------------------------------
// Small helper kernels
// ---------------------------------------------------------------------------
__global__ void cvt_bf16(const float* __restrict__ a, __bf16* __restrict__ o, int n) {
    int i = blockIdx.x * blockDim.x + threadIdx.x;
    if (i < n) o[i] = (__bf16)a[i];
}

__global__ void gin_agg(const float* __restrict__ x, const long long* __restrict__ ei,
                        float* __restrict__ aggr, int E, int D) {
    const long long src = ei[blockIdx.x];
    const long long dst = ei[(size_t)E + blockIdx.x];
    atomicAdd(&aggr[(size_t)dst * D + threadIdx.x], x[(size_t)src * D + threadIdx.x]);
}

__global__ void vec_add_bf(const float* __restrict__ a, const float* __restrict__ b,
                           __bf16* __restrict__ o, int n) {
    int i = blockIdx.x * blockDim.x + threadIdx.x;
    if (i < n) o[i] = (__bf16)(a[i] + b[i]);
}

__global__ void vec_add_dual(const float* __restrict__ a, const float* __restrict__ b,
                             float* __restrict__ of, __bf16* __restrict__ ob, int n) {
    int i = blockIdx.x * blockDim.x + threadIdx.x;
    if (i < n) { float s = a[i] + b[i]; of[i] = s; ob[i] = (__bf16)s; }
}

__global__ void row_sq(const float* __restrict__ p, float* __restrict__ a2, int D) {
    __shared__ float red[256];
    const int row = blockIdx.x;
    float v = p[(size_t)row * D + threadIdx.x];
    red[threadIdx.x] = v * v;
    __syncthreads();
    for (int s = 128; s > 0; s >>= 1) {
        if (threadIdx.x < s) red[threadIdx.x] += red[threadIdx.x + s];
        __syncthreads();
    }
    if (threadIdx.x == 0) a2[row] = red[0];
}

// Coalesced BN stats: phase 1 partial sums (col = tid), phase 2 finalize.
__global__ void bn_partial(const float* __restrict__ p1, const float* __restrict__ p2,
                           float* __restrict__ accSum, float* __restrict__ accSq,
                           int rowsPerBlock, int D) {
    const int col = threadIdx.x;
    const int r0  = blockIdx.x * rowsPerBlock;
    float s = 0.f, s2 = 0.f;
    for (int r = r0; r < r0 + rowsPerBlock; ++r) {
        float v = p1[(size_t)r * D + col];
        if (p2) v += p2[(size_t)r * D + col];
        s += v; s2 += v * v;
    }
    atomicAdd(&accSum[col], s);
    atomicAdd(&accSq[col], s2);
}

__global__ void bn_final(const float* __restrict__ accSum, const float* __restrict__ accSq,
                         float* __restrict__ mean, float* __restrict__ rstd, int M) {
    const int c = threadIdx.x;
    float mu  = accSum[c] / (float)M;
    float var = accSq[c] / (float)M - mu * mu;
    mean[c] = mu;
    rstd[c] = rsqrtf(var + 1e-5f);
}

__global__ void bn_apply(const float* __restrict__ p1, const float* __restrict__ p2,
                         const float* __restrict__ mean, const float* __restrict__ rstd,
                         const float* __restrict__ g, const float* __restrict__ b,
                         float* __restrict__ o, int n, int D) {
    int i = blockIdx.x * blockDim.x + threadIdx.x;
    if (i >= n) return;
    int c = i % D;
    float v = p1[i];
    if (p2) v += p2[i];
    o[i] = (v - mean[c]) * rstd[c] * g[c] + b[c];
}

// ---------------------------------------------------------------------------
// Host launcher
// ---------------------------------------------------------------------------
extern "C" void kernel_launch(void* const* d_in, const int* in_sizes, int n_in,
                              void* d_out, int out_size, void* d_ws, size_t ws_size,
                              hipStream_t stream) {
    constexpr int B = 2, N = 2048, D = 256, H = 8, HD = 32, DFF = 512;
    constexpr int BN = B * N; // 4096 rows
    const int E = in_sizes[24] / 2;

    const float* x    = (const float*)d_in[0];
    const float* pos  = (const float*)d_in[1];
    const float* gw1  = (const float*)d_in[2];  const float* gb1 = (const float*)d_in[3];
    const float* gw2  = (const float*)d_in[4];  const float* gb2 = (const float*)d_in[5];
    const float* wq   = (const float*)d_in[6];  const float* bq  = (const float*)d_in[7];
    const float* wk   = (const float*)d_in[8];  const float* bk  = (const float*)d_in[9];
    const float* wv   = (const float*)d_in[10]; const float* bv  = (const float*)d_in[11];
    const float* wo   = (const float*)d_in[12]; const float* bo  = (const float*)d_in[13];
    const float* bnlg = (const float*)d_in[14]; const float* bnlb = (const float*)d_in[15];
    const float* bnag = (const float*)d_in[16]; const float* bnab = (const float*)d_in[17];
    const float* bn2g = (const float*)d_in[18]; const float* bn2b = (const float*)d_in[19];
    const float* fw1  = (const float*)d_in[20]; const float* fb1 = (const float*)d_in[21];
    const float* fw2  = (const float*)d_in[22]; const float* fb2 = (const float*)d_in[23];
    const long long* ei = (const long long*)d_in[24];

    char* wsp = (char*)d_ws;
    auto alloc = [&](size_t bytes) -> void* {
        void* p = wsp;
        wsp += (bytes + 255) & ~(size_t)255;
        return p;
    };
    const size_t S  = (size_t)BN * D;       // 1M elements
    const size_t SW = (size_t)D * D;        // 64K weight elements

    float*  aggr   = (float*)alloc(S * 4);
    float*  hll    = (float*)alloc(S * 4);
    float*  hlocal = (float*)alloc(S * 4);
    float*  attnp  = (float*)alloc(S * 4);
    float*  hattn  = (float*)alloc(S * 4);
    float*  hbuf   = (float*)alloc(S * 4);
    float*  ff2    = (float*)alloc(S * 4);
    float*  biasW  = (float*)alloc((size_t)B * N * N * 4); // 33.5MB, L2 resident
    float*  a2     = (float*)alloc((size_t)BN * 4);
    float*  accs   = (float*)alloc(2 * D * 4);   // [sum | sumsq]
    float*  accq   = accs + D;
    float*  meanb  = (float*)alloc(D * 4);
    float*  rstdb  = (float*)alloc(D * 4);
    __bf16* xb     = (__bf16*)alloc(S * 2);
    __bf16* posb   = (__bf16*)alloc(S * 2);
    __bf16* zb     = (__bf16*)alloc(S * 2);
    __bf16* h1b    = (__bf16*)alloc(S * 2);
    __bf16* hbufb  = (__bf16*)alloc(S * 2);
    __bf16* ffh1b  = (__bf16*)alloc((size_t)BN * DFF * 2);
    __bf16* attnb  = (__bf16*)alloc(S * 2);
    __bf16* qb     = (__bf16*)alloc(S * 2);
    __bf16* kbuf   = (__bf16*)alloc(S * 2);
    __bf16* vbuf   = (__bf16*)alloc(S * 2);
    __bf16* gw1b   = (__bf16*)alloc(SW * 2);
    __bf16* gw2b   = (__bf16*)alloc(SW * 2);
    __bf16* wqb    = (__bf16*)alloc(SW * 2);
    __bf16* wkb    = (__bf16*)alloc(SW * 2);
    __bf16* wvb    = (__bf16*)alloc(SW * 2);
    __bf16* wob    = (__bf16*)alloc(SW * 2);
    __bf16* fw1b   = (__bf16*)alloc((size_t)DFF * D * 2);
    __bf16* fw2b   = (__bf16*)alloc((size_t)D * DFF * 2);

    const dim3 blk256(256);
    const int elemBlocks = (int)(S / 256);
    const int wBlocks    = (int)(SW / 256);
    constexpr int BN_ROWS = 128;                 // rows per bn_partial block
    const int bnBlocks = BN / BN_ROWS;

    auto run_bn = [&](const float* p1, const float* p2, const float* g,
                      const float* bvec, float* o) {
        hipMemsetAsync(accs, 0, 2 * D * 4, stream);
        bn_partial<<<bnBlocks, blk256, 0, stream>>>(p1, p2, accs, accq, BN_ROWS, D);
        bn_final<<<1, blk256, 0, stream>>>(accs, accq, meanb, rstdb, BN);
        bn_apply<<<elemBlocks, blk256, 0, stream>>>(p1, p2, meanb, rstdb, g, bvec,
                                                    o, (int)S, D);
    };

    // ---- one-time bf16 conversions ----
    cvt_bf16<<<elemBlocks, blk256, 0, stream>>>(x, xb, (int)S);
    cvt_bf16<<<elemBlocks, blk256, 0, stream>>>(pos, posb, (int)S);
    cvt_bf16<<<wBlocks, blk256, 0, stream>>>(gw1, gw1b, (int)SW);
    cvt_bf16<<<wBlocks, blk256, 0, stream>>>(gw2, gw2b, (int)SW);
    cvt_bf16<<<wBlocks, blk256, 0, stream>>>(wq, wqb, (int)SW);
    cvt_bf16<<<wBlocks, blk256, 0, stream>>>(wk, wkb, (int)SW);
    cvt_bf16<<<wBlocks, blk256, 0, stream>>>(wv, wvb, (int)SW);
    cvt_bf16<<<wBlocks, blk256, 0, stream>>>(wo, wob, (int)SW);
    cvt_bf16<<<2 * wBlocks, blk256, 0, stream>>>(fw1, fw1b, (int)(2 * SW));
    cvt_bf16<<<2 * wBlocks, blk256, 0, stream>>>(fw2, fw2b, (int)(2 * SW));

    // ---- local branch: GIN ----
    hipMemsetAsync(aggr, 0, S * 4, stream);
    gin_agg<<<E, D, 0, stream>>>(x, ei, aggr, E, D);
    vec_add_bf<<<elemBlocks, blk256, 0, stream>>>(x, aggr, zb, (int)S);
    gemm_nt<1, D, D><<<dim3(BN / 128, D / 64, 1), blk256, 0, stream>>>(
        zb, gw1b, gb1, nullptr, h1b, nullptr, 1.f, 0, 0, 0);
    gemm_nt<0, D, D><<<dim3(BN / 128, D / 64, 1), blk256, 0, stream>>>(
        h1b, gw2b, gb2, hll, nullptr, nullptr, 1.f, 0, 0, 0);
    run_bn(x, hll, bnlg, bnlb, hlocal);

    // ---- attention bias: 2+1e-24 - pdist(pos_enc) ----
    row_sq<<<BN, blk256, 0, stream>>>(pos, a2, D);
    gemm_nt<3, N, D><<<dim3(N / 128, N / 64, B), blk256, 0, stream>>>(
        posb, posb, nullptr, biasW, nullptr, a2, 1.f,
        (long)N * D, (long)N * D, (long)N * N);

    // ---- QKV projections (bf16 outputs; Q pre-scaled by 1/sqrt(hd)) ----
    const float qscale = 0.17677669529663687f; // 1/sqrt(32)
    gemm_nt<2, D, D><<<dim3(BN / 128, D / 64, 1), blk256, 0, stream>>>(
        xb, wqb, bq, nullptr, qb, nullptr, qscale, 0, 0, 0);
    gemm_nt<2, D, D><<<dim3(BN / 128, D / 64, 1), blk256, 0, stream>>>(
        xb, wkb, bk, nullptr, kbuf, nullptr, 1.f, 0, 0, 0);
    gemm_nt<2, D, D><<<dim3(BN / 128, D / 64, 1), blk256, 0, stream>>>(
        xb, wvb, bv, nullptr, vbuf, nullptr, 1.f, 0, 0, 0);

    // ---- flash attention ----
    flash_attn<N, D, H, HD><<<dim3(N / 16, B * H), dim3(32), 0, stream>>>(
        qb, kbuf, vbuf, biasW, attnb);

    // ---- output projection + BN ----
    gemm_nt<0, D, D><<<dim3(BN / 128, D / 64, 1), blk256, 0, stream>>>(
        attnb, wob, bo, attnp, nullptr, nullptr, 1.f, 0, 0, 0);
    run_bn(x, attnp, bnag, bnab, hattn);

    // ---- combine + FFN + final BN ----
    vec_add_dual<<<elemBlocks, blk256, 0, stream>>>(hlocal, hattn, hbuf, hbufb, (int)S);
    gemm_nt<1, DFF, D><<<dim3(BN / 128, DFF / 64, 1), blk256, 0, stream>>>(
        hbufb, fw1b, fb1, nullptr, ffh1b, nullptr, 1.f, 0, 0, 0);
    gemm_nt<0, D, DFF><<<dim3(BN / 128, D / 64, 1), blk256, 0, stream>>>(
        ffh1b, fw2b, fb2, ff2, nullptr, nullptr, 1.f, 0, 0, 0);
    run_bn(hbuf, ff2, bn2g, bn2b, (float*)d_out);
}
